// MultiHeadAttentionBlock_23579370455489
// MI455X (gfx1250) — compile-verified, matrix operand order checked by reading, not run on hardware
//
#include <hip/hip_runtime.h>
#include <hip/hip_bf16.h>

// MHA block: B=4, S=2048, D=1024, H=16, DK=64. All matmuls on
// v_wmma_f32_16x16x32_bf16 (fp32 accum). Causal mask applied analytically,
// only on the diagonal key-tile. Attention K/V tiles staged with the
// Tensor Data Mover (tensor_load_to_lds, TENSORcnt); final-projection A tile
// staged with global_load_async_to_lds_b128 (ASYNCcnt).

#define S_LEN 2048
#define D_DIM 1024
#define H_N   16
#define DK_N  64

typedef __attribute__((ext_vector_type(16))) __bf16 v16bf;
typedef __attribute__((ext_vector_type(8)))  float  v8f;
typedef __attribute__((ext_vector_type(4)))  unsigned int u32x4;
typedef __attribute__((ext_vector_type(4)))  int i32x4;
typedef __attribute__((ext_vector_type(8)))  int i32x8;

union Frag16 { v16bf v; uint4 q[2]; };

#define WMMA_BF16(A, B, C) \
  __builtin_amdgcn_wmma_f32_16x16x32_bf16(false, (A), false, (B), (short)0, (C), false, false)

__device__ __forceinline__ unsigned short f2bf(float f) {
  unsigned int u = __float_as_uint(f);
  u += 0x7FFFu + ((u >> 16) & 1u);   // round-to-nearest-even
  return (unsigned short)(u >> 16);
}

// Pack two fp32 -> two bf16 in one dword: per-value round bias + v_perm_b32.
__device__ __forceinline__ unsigned pack2bf(float a, float b) {
  unsigned ua = __float_as_uint(a) + 0x8000u;
  unsigned ub = __float_as_uint(b) + 0x8000u;
  return __builtin_amdgcn_perm(ub, ua, 0x07060302u);  // {hi16(b), hi16(a)}
}

// Async copy 16B/lane from global to LDS (gfx1250, tracked by ASYNCcnt).
__device__ __forceinline__ void async_cp16(const unsigned short* gsrc,
                                           unsigned short* ldst) {
  const unsigned lds_off = (unsigned)(unsigned long long)ldst;  // addr[31:0] = LDS offset
  asm volatile("global_load_async_to_lds_b128 %0, %1, off"
               :: "v"(lds_off), "v"((unsigned long long)gsrc)
               : "memory");
}
__device__ __forceinline__ void async_wait0() {
  asm volatile("s_wait_asynccnt 0" ::: "memory");
}

// ---- Tensor Data Mover: 2D 16-bit tile load, global -> padded LDS ----
// D# group 0: count=1 | lds_addr | global_addr[56:0] | type=2 (ISA 8.3)
__device__ __forceinline__ u32x4 tdm_g0(unsigned lds_off, const void* gaddr) {
  const unsigned long long ga = (unsigned long long)gaddr;
  u32x4 g;
  g.x = 1u;                                   // count = 1 valid descriptor
  g.y = lds_off;                              // bits[63:32]: LDS byte address
  g.z = (unsigned)ga;                         // bits[95:64]: global addr lo
  g.w = (unsigned)(ga >> 32) | (2u << 30);    // addr[56:32] | type=2 ("image")
  return g;
}
// D# group 1 (ISA 8.4): data_size=2B; LDS pad: every 32 DWORDs (one 64-elem
// row) insert 4 DWORDs (8 elems) -> stride-72 rows. 64x64 tile.
__device__ __forceinline__ i32x8 tdm_g1_64x64(unsigned long long stride0) {
  const unsigned td0 = 1u << 20, td1 = 1u << 20;  // generous OOB extents
  i32x8 g;
  g[0] = (1 << 16)        // data_size = 1 -> 2 bytes
       | (1 << 20)        // pad_enable
       | (4 << 22)        // pad_interval code 4 -> 32 DWORDs
       | (3 << 25);       // pad_amount  code 3 -> 4 DWORDs
  g[1] = (int)((td0 & 0xFFFFu) << 16);                       // tensor_dim0 lo
  g[2] = (int)(((td0 >> 16) & 0xFFFFu) | ((td1 & 0xFFFFu) << 16));
  g[3] = (int)(((td1 >> 16) & 0xFFFFu) | (64u << 16));       // tile_dim0 = 64
  g[4] = 64;                                                 // tile_dim1 = 64
  g[5] = (int)(unsigned)(stride0 & 0xFFFFFFFFull);           // dim0 stride lo
  g[6] = (int)(unsigned)((stride0 >> 32) & 0xFFFFull);       // stride hi | dim1_stride lo
  g[7] = 0;
  return g;
}
__device__ __forceinline__ void tdm_load_tile(unsigned short* ldst,
                                              const unsigned short* gsrc,
                                              unsigned long long stride_elems) {
  const u32x4 g0 = tdm_g0((unsigned)(unsigned long long)ldst, gsrc);
  const i32x8 g1 = tdm_g1_64x64(stride_elems);
  const i32x4 gz = {};
#if defined(__clang_major__) && __clang_major__ >= 23
  const i32x8 gz8 = {};
  __builtin_amdgcn_tensor_load_to_lds(g0, g1, gz, gz, gz8, 0);
#else
  __builtin_amdgcn_tensor_load_to_lds(g0, g1, gz, gz, 0);
#endif
}

// Load a 16x32 bf16 WMMA fragment (A-layout; B uses the same pattern on B^T):
// lane = {half = lane>>4, idx = lane&15}; row = base+idx; two 16B chunks at
// K = half*8 and K = 16 + half*8 (per CDNA5 ISA 7.12.2 16-bit A layout).
__device__ __forceinline__ v16bf load_frag16(const unsigned short* p, int stride,
                                             int row, int kBase) {
  const int lane = threadIdx.x & 31;
  const int hl = lane >> 4, idx = lane & 15;
  const unsigned short* b = p + (size_t)(row + idx) * stride + (kBase + hl * 8);
  Frag16 f;
  f.q[0] = *(const uint4*)(b);
  f.q[1] = *(const uint4*)(b + 16);
  return f.v;
}

// Y[m,n] = sum_k A[m,k]*W[n,k] + bias[n];  M=8192, N=K=1024.
// MODE 0: A fp32 -> Qh bf16 [B,H,S,DK], scaled 1/sqrt(DK)
// MODE 1: A fp32 -> Kh bf16 [B,H,S,DK]
// MODE 2: A fp32 -> Vt bf16 [B,H,DK,S]   (transposed for PV B-fragments)
// MODE 3: A bf16 -> fp32 [M,N] (final projection; A staged via async-to-LDS)
template <int MODE>
__global__ __launch_bounds__(256) void gemm_proj(const void* __restrict__ Aptr,
                                                 const float* __restrict__ W,
                                                 const float* __restrict__ bias,
                                                 void* __restrict__ outp) {
  __shared__ unsigned short As[128 * 40];  // 128 x 32, stride 40 (pad)
  __shared__ unsigned short Ws[128 * 40];
  const int t = threadIdx.x;
  const int rb = blockIdx.x * 128;
  const int cb = blockIdx.y * 128;
  const int wave = t >> 5;
  const int wm = wave >> 2, wn = wave & 3;  // 2 (M) x 4 (N) waves, 64x32 each

  v8f zero = {};
  v8f acc[4][2];
  for (int i = 0; i < 4; ++i)
    for (int j = 0; j < 2; ++j) acc[i][j] = zero;

  for (int kb = 0; kb < D_DIM; kb += 32) {
    __syncthreads();
    {  // stage W tile (fp32 -> bf16 via v_perm packing): 128 n-rows x 32 k
      const int r = t >> 1, c = (t & 1) * 16;
      const float* src = W + (size_t)(cb + r) * D_DIM + kb + c;
      if (kb + 32 < D_DIM) __builtin_prefetch(src + 32, 0, 0);
      const float4* s4 = (const float4*)src;
      alignas(16) unsigned pk[8];
#pragma unroll
      for (int i = 0; i < 4; ++i) {
        float4 f4 = s4[i];
        pk[i * 2 + 0] = pack2bf(f4.x, f4.y);
        pk[i * 2 + 1] = pack2bf(f4.z, f4.w);
      }
      *(uint4*)(Ws + r * 40 + c)     = *(const uint4*)(pk);
      *(uint4*)(Ws + r * 40 + c + 8) = *(const uint4*)(pk + 4);
    }
    if constexpr (MODE == 3) {  // A already bf16: direct async mem->LDS copy
      const unsigned short* A = (const unsigned short*)Aptr;
#pragma unroll
      for (int i = 0; i < 2; ++i) {
        const int c = t + i * 256;  // 512 chunks of 8 elems
        const int r = c >> 2, cc = (c & 3) * 8;
        async_cp16(A + (size_t)(rb + r) * D_DIM + kb + cc, As + r * 40 + cc);
      }
      async_wait0();
    } else {  // A fp32 -> bf16
      const float* A = (const float*)Aptr;
      const int r = t >> 1, c = (t & 1) * 16;
      const float* src = A + (size_t)(rb + r) * D_DIM + kb + c;
      if (kb + 32 < D_DIM) __builtin_prefetch(src + 32, 0, 0);
      const float4* s4 = (const float4*)src;
      alignas(16) unsigned pk[8];
#pragma unroll
      for (int i = 0; i < 4; ++i) {
        float4 f4 = s4[i];
        pk[i * 2 + 0] = pack2bf(f4.x, f4.y);
        pk[i * 2 + 1] = pack2bf(f4.z, f4.w);
      }
      *(uint4*)(As + r * 40 + c)     = *(const uint4*)(pk);
      *(uint4*)(As + r * 40 + c + 8) = *(const uint4*)(pk + 4);
    }
    __syncthreads();

    v16bf af[4], bfr[2];
#pragma unroll
    for (int fm = 0; fm < 4; ++fm) af[fm] = load_frag16(As, 40, wm * 64 + fm * 16, 0);
#pragma unroll
    for (int fn = 0; fn < 2; ++fn) bfr[fn] = load_frag16(Ws, 40, wn * 32 + fn * 16, 0);
#pragma unroll
    for (int fm = 0; fm < 4; ++fm)
#pragma unroll
      for (int fn = 0; fn < 2; ++fn)
        acc[fm][fn] = WMMA_BF16(af[fm], bfr[fn], acc[fm][fn]);
  }

  // Epilogue. C layout: lanes 0-15 -> M = r, lanes 16-31 -> M = r+8; N = lane&15.
  const int lane = t & 31;
  const int hl = lane >> 4, nn = lane & 15;
#pragma unroll
  for (int fm = 0; fm < 4; ++fm)
#pragma unroll
    for (int fn = 0; fn < 2; ++fn)
#pragma unroll
      for (int r = 0; r < 8; ++r) {
        const int m_abs = rb + wm * 64 + fm * 16 + r + hl * 8;
        const int n_abs = cb + wn * 32 + fn * 16 + nn;
        float v = acc[fm][fn][r] + bias[n_abs];
        if constexpr (MODE == 3) {
          ((float*)outp)[(size_t)m_abs * D_DIM + n_abs] = v;
        } else {
          unsigned short* O = (unsigned short*)outp;
          const int b = m_abs >> 11, s = m_abs & (S_LEN - 1);
          const int h = n_abs >> 6, dk = n_abs & (DK_N - 1);
          if constexpr (MODE == 0) v *= 0.125f;  // 1/sqrt(64) folded into Q
          if constexpr (MODE == 2)
            O[(((size_t)b * H_N + h) * DK_N + dk) * S_LEN + s] = f2bf(v);
          else
            O[(((size_t)b * H_N + h) * S_LEN + s) * DK_N + dk] = f2bf(v);
        }
      }
}

// One 64-key tile of flash attention for one wave (16 q rows).
// MASKED is true only for the diagonal tile.
template <bool MASKED>
__device__ __forceinline__ void attn_tile(
    int kvb, int qbase, int hl, int nn, int t,
    const unsigned short* __restrict__ Kp, const unsigned short* __restrict__ Vp,
    unsigned short* Ks, unsigned short* Vs, unsigned short* pw,
    const v16bf& qa0, const v16bf& qa1,
    v8f o[4], float mrow[8], float lrow[8]) {
  __syncthreads();  // previous tile's LDS reads complete
  // TDM-stage K tile (64 kv x 64 dk) and V^T tile (64 dk x 64 kv):
  // one descriptor each, hardware applies the stride-72 LDS padding.
  if (t < 32) {
    tdm_load_tile(Ks, Kp + (size_t)kvb * DK_N, DK_N);   // rows = kv, stride 64
    tdm_load_tile(Vs, Vp + kvb, S_LEN);                 // rows = dk, stride 2048
    __builtin_amdgcn_s_wait_tensorcnt(0);
  }
  __syncthreads();

  // S = Q K^T : 16 x 64 scores per wave (scale folded into Q). 8 WMMAs.
  v8f s[4];
#pragma unroll
  for (int c = 0; c < 4; ++c) {
    v8f sc = {};
    sc = WMMA_BF16(qa0, load_frag16(Ks, 72, c * 16, 0), sc);
    sc = WMMA_BF16(qa1, load_frag16(Ks, 72, c * 16, 32), sc);
    s[c] = sc;
  }

  // Online softmax; row reductions stay within the 16-lane halves.
  float alpha[8];
#pragma unroll
  for (int r = 0; r < 8; ++r) {
    const int qrow = qbase + r + hl * 8;
    float e[4];
#pragma unroll
    for (int c = 0; c < 4; ++c) {
      e[c] = s[c][r];
      if (MASKED) {
        if (kvb + c * 16 + nn > qrow) e[c] = -1.0e30f;
      }
    }
    float mx = fmaxf(fmaxf(e[0], e[1]), fmaxf(e[2], e[3]));
    mx = fmaxf(mx, __shfl_xor(mx, 1, 32));
    mx = fmaxf(mx, __shfl_xor(mx, 2, 32));
    mx = fmaxf(mx, __shfl_xor(mx, 4, 32));
    mx = fmaxf(mx, __shfl_xor(mx, 8, 32));
    const float mnew = fmaxf(mrow[r], mx);
    float p[4], sum = 0.0f;
#pragma unroll
    for (int c = 0; c < 4; ++c) { p[c] = __expf(e[c] - mnew); sum += p[c]; }
    sum += __shfl_xor(sum, 1, 32);
    sum += __shfl_xor(sum, 2, 32);
    sum += __shfl_xor(sum, 4, 32);
    sum += __shfl_xor(sum, 8, 32);
    alpha[r] = __expf(mrow[r] - mnew);
    lrow[r] = lrow[r] * alpha[r] + sum;
    mrow[r] = mnew;
    const int prow = r + hl * 8;  // C-layout row
#pragma unroll
    for (int c = 0; c < 4; ++c) pw[prow * 72 + c * 16 + nn] = f2bf(p[c]);
  }
#pragma unroll
  for (int d = 0; d < 4; ++d)
#pragma unroll
    for (int r = 0; r < 8; ++r) o[d][r] *= alpha[r];

  __syncthreads();  // P visible in A-layout via LDS round-trip

  // O += P V : two 16x32 A-frags of P, 4 dk chunks of V^T. 8 WMMAs.
  const v16bf pf0 = load_frag16(pw, 72, 0, 0);
  const v16bf pf1 = load_frag16(pw, 72, 0, 32);
#pragma unroll
  for (int d = 0; d < 4; ++d) {
    o[d] = WMMA_BF16(pf0, load_frag16(Vs, 72, d * 16, 0), o[d]);
    o[d] = WMMA_BF16(pf1, load_frag16(Vs, 72, d * 16, 32), o[d]);
  }
}

// Flash attention: block = (bh, 64 q-rows), 4 waves x 16 rows, 64-key tiles.
__global__ __launch_bounds__(128) void attn_fwd(const unsigned short* __restrict__ Qh,
                                                const unsigned short* __restrict__ Kh,
                                                const unsigned short* __restrict__ Vt,
                                                unsigned short* __restrict__ Out) {
  __shared__ unsigned short Ks[64 * 72];       // 64 kv x 64 dk
  __shared__ unsigned short Vs[64 * 72];       // 64 dk x 64 kv  (V^T tile)
  __shared__ unsigned short Ps[4 * 16 * 72];   // per-wave 16 x 64 probs
  const int t = threadIdx.x;
  const int bh = blockIdx.x;
  const int qb = blockIdx.y * 64;
  const int wave = t >> 5;
  const int lane = t & 31;
  const int hl = lane >> 4, nn = lane & 15;
  const int qbase = qb + wave * 16;

  const unsigned short* Qp = Qh + (size_t)bh * S_LEN * DK_N;
  const unsigned short* Kp = Kh + (size_t)bh * S_LEN * DK_N;
  const unsigned short* Vp = Vt + (size_t)bh * DK_N * S_LEN;

  const v16bf qa0 = load_frag16(Qp, DK_N, qbase, 0);    // K = 0..31
  const v16bf qa1 = load_frag16(Qp, DK_N, qbase, 32);   // K = 32..63

  v8f zero = {};
  v8f o[4];
  for (int d = 0; d < 4; ++d) o[d] = zero;
  float mrow[8], lrow[8];
  for (int r = 0; r < 8; ++r) { mrow[r] = -3.0e38f; lrow[r] = 0.0f; }

  unsigned short* pw = Ps + wave * 16 * 72;

  // Fully-unmasked tiles below the diagonal (uniform count per block).
  const int nfull = blockIdx.y;
  for (int j = 0; j < nfull; ++j)
    attn_tile<false>(j * 64, qbase, hl, nn, t, Kp, Vp, Ks, Vs, pw,
                     qa0, qa1, o, mrow, lrow);
  // Diagonal tile with causal mask.
  attn_tile<true>(nfull * 64, qbase, hl, nn, t, Kp, Vp, Ks, Vs, pw,
                  qa0, qa1, o, mrow, lrow);

  // Epilogue: normalize and write bf16 [B, S, D] with heads re-interleaved.
  const size_t b = (size_t)(bh >> 4);
  const int h = bh & (H_N - 1);
#pragma unroll
  for (int r = 0; r < 8; ++r) {
    const float inv = 1.0f / lrow[r];
    const int qrow = qbase + r + hl * 8;
#pragma unroll
    for (int d = 0; d < 4; ++d)
      Out[(b * S_LEN + qrow) * D_DIM + (size_t)h * DK_N + d * 16 + nn] =
          f2bf(o[d][r] * inv);
  }
}

extern "C" void kernel_launch(void* const* d_in, const int* in_sizes, int n_in,
                              void* d_out, int out_size, void* d_ws, size_t ws_size,
                              hipStream_t stream) {
  (void)in_sizes; (void)n_in; (void)out_size; (void)ws_size;
  const float* query = (const float*)d_in[0];
  const float* keyin = (const float*)d_in[1];
  const float* value = (const float*)d_in[2];
  // d_in[3] = mask (unused; causal handled analytically)
  const float* Wq = (const float*)d_in[4];
  const float* bq = (const float*)d_in[5];
  const float* Wk = (const float*)d_in[6];
  const float* bk = (const float*)d_in[7];
  const float* Wv = (const float*)d_in[8];
  const float* bv = (const float*)d_in[9];
  const float* Wo = (const float*)d_in[10];
  const float* bo = (const float*)d_in[11];

  const size_t nBHSD = (size_t)4 * H_N * S_LEN * DK_N;  // 8,388,608 elems (bf16)
  unsigned short* Qh  = (unsigned short*)d_ws;
  unsigned short* Khd = Qh + nBHSD;
  unsigned short* Vtd = Khd + nBHSD;
  unsigned short* Ah  = Vtd + nBHSD;   // total 64 MB of d_ws

  dim3 gg(64, 8);  // 8192/128 x 1024/128
  gemm_proj<0><<<gg, 256, 0, stream>>>(query, Wq, bq, Qh);
  gemm_proj<1><<<gg, 256, 0, stream>>>(keyin, Wk, bk, Khd);
  gemm_proj<2><<<gg, 256, 0, stream>>>(value, Wv, bv, Vtd);
  attn_fwd<<<dim3(64, 32), 128, 0, stream>>>(Qh, Khd, Vtd, Ah);
  gemm_proj<3><<<gg, 256, 0, stream>>>(Ah, Wo, bo, d_out);
}
